// RGAST_38800734552490
// MI455X (gfx1250) — matrix operand order repaired and checked by reading, not compiled, not run on hardware
//
#include <hip/hip_runtime.h>
#include <hip/hip_bf16.h>

typedef __attribute__((ext_vector_type(16))) _Float16 v16h;
typedef __attribute__((ext_vector_type(8)))  float    v8f;
typedef __attribute__((ext_vector_type(4)))  float    f32x4;
typedef __attribute__((ext_vector_type(4)))  int      i32x4;

#define TPB 256

// ---------------- helpers ----------------
__device__ __forceinline__ unsigned fenc(float x) {
    unsigned u = __float_as_uint(x);
    return (u & 0x80000000u) ? ~u : (u | 0x80000000u);
}
__device__ __forceinline__ float fdec(unsigned e) {
    return (e & 0x80000000u) ? __uint_as_float(e ^ 0x80000000u)
                             : __uint_as_float(~e);
}

// ---------------- conversion ----------------
__global__ void k_f32_to_f16(const float* __restrict__ in, _Float16* __restrict__ out, int n) {
    int i = blockIdx.x * blockDim.x + threadIdx.x;
    if (i < n) out[i] = (_Float16)in[i];
}

// Pack row-major W [Kdim x Nout] (f32) into per-lane WMMA B-fragments (f16).
// B-fragment layout (16x16x32 f16): lane = hw*16+n ; element i (0..15) holds
// B[k = ks*32 + hw*16 + i][col = ct*16 + n].
__global__ void k_pack_bfrag(const float* __restrict__ W, _Float16* __restrict__ frag,
                             int Kdim, int Nout) {
    int tid = blockIdx.x * blockDim.x + threadIdx.x;
    int KS = Kdim >> 5, CT = Nout >> 4;
    if (tid >= CT * KS * 32) return;
    int lane = tid & 31;
    int fid  = tid >> 5;          // ct*KS + ks
    int ct = fid / KS, ks = fid - ct * KS;
    int n = lane & 15, hw = lane >> 4;
    _Float16* dst = frag + (size_t)tid * 16;
    const float* wp = W + (size_t)(ks * 32 + hw * 16) * Nout + ct * 16 + n;
    #pragma unroll
    for (int i = 0; i < 16; ++i) dst[i] = (_Float16)wp[(size_t)i * Nout];
}

// Tiled WMMA GEMM. Block = 8 waves; all waves share one column group of
// NT 16-wide tiles whose B-fragments are staged in LDS once; wave w computes
// rows [tm*16, tm*16+16) x NT tiles. KDIM/CTT compile-time -> unrolled K loop,
// immediate-offset loads/stores. B fragments are read into independent
// registers per k-step so the ds_loads batch and the 4 WMMAs overlap.
template <int KDIM, int CTT, int NT>
__global__ void k_wmma_gemm_t(const _Float16* __restrict__ A,
                              const _Float16* __restrict__ Bfrag,
                              float* __restrict__ C,
                              const float* __restrict__ bias, int Mt) {
    constexpr int KS   = KDIM / 32;
    constexpr int CG   = CTT / NT;       // column groups
    constexpr int NOUT = CTT * 16;
    __shared__ _Float16 lb[NT * KS * 512];   // NT*KS fragments, 512 halves each

    int tid  = threadIdx.x;
    int wid  = tid >> 5, lane = tid & 31;
    int cg   = blockIdx.x % CG;
    int rg   = blockIdx.x / CG;

    // ---- stage B fragments for this column group into LDS (16B chunks)
    {
        const i32x4* gsrc = (const i32x4*)(Bfrag + (size_t)cg * NT * KS * 512);
        i32x4* ldst = (i32x4*)lb;
        constexpr int NCHUNK = NT * KS * 64;      // 16B chunks
        #pragma unroll
        for (int i = 0; i < (NCHUNK + TPB - 1) / TPB; ++i) {
            int idx = tid + i * TPB;
            if (idx < NCHUNK) ldst[idx] = gsrc[idx];
        }
    }
    __syncthreads();

    int tm = rg * 8 + wid;
    // wave-uniform guard via SGPR so EXEC stays all-1s around WMMA
    if (__builtin_amdgcn_readfirstlane(tm) < Mt) {
        int row   = tm * 16 + (lane & 15);
        int klane = (lane >> 4) << 4;
        const _Float16* abase = A + (size_t)row * KDIM + klane;
        const v16h* bl = (const v16h*)lb;

        v8f acc[NT];
        #pragma unroll
        for (int nt = 0; nt < NT; ++nt) acc[nt] = (v8f){};

        #pragma unroll
        for (int ks = 0; ks < KS; ++ks) {
            union { v16h h; f32x4 q[2]; } a;
            a.q[0] = *(const f32x4*)(abase);
            a.q[1] = *(const f32x4*)(abase + 8);
            __builtin_prefetch(abase + 32, 0, 1);   // global_prefetch_b8

            v16h bfr[NT];                           // independent regs -> batched ds_loads
            #pragma unroll
            for (int nt = 0; nt < NT; ++nt)
                bfr[nt] = bl[(nt * KS + ks) * 32 + lane];
            #pragma unroll
            for (int nt = 0; nt < NT; ++nt)
                acc[nt] = __builtin_amdgcn_wmma_f32_16x16x32_f16(
                    false, a.h, false, bfr[nt], (short)0, acc[nt], false, false);
            abase += 32;
        }

        // C layout: VGPR j -> row (lane<16 ? j : 8+j), col = lane&15
        int colbase = cg * NT * 16 + (lane & 15);
        int mrow    = tm * 16 + ((lane < 16) ? 0 : 8);
        float* crow = C + (size_t)mrow * NOUT + colbase;
        #pragma unroll
        for (int nt = 0; nt < NT; ++nt) {
            float badd = bias ? bias[colbase + nt * 16] : 0.0f;
            #pragma unroll
            for (int j = 0; j < 8; ++j)
                crow[(size_t)j * NOUT + nt * 16] = acc[nt][j] + badd;
        }
    }
}

// Per-(relation,node) attention scalars: qn = xw.Q , kn = xw.K
__global__ void k_node_qk(const float* __restrict__ xw, const float* __restrict__ Q,
                          const float* __restrict__ Kv, float* __restrict__ qn,
                          float* __restrict__ kn, int NR, int N, int O) {
    int idx = blockIdx.x * blockDim.x + threadIdx.x;
    if (idx >= NR) return;
    int r = idx / N;
    const float* x = xw + (size_t)idx * O;
    const float* q = Q + (size_t)r * O;
    const float* k = Kv + (size_t)r * O;
    float sq = 0.f, sk = 0.f;
    for (int i = 0; i < O; ++i) { float v = x[i]; sq += v * q[i]; sk += v * k[i]; }
    qn[idx] = sq; kn[idx] = sk;
}

__global__ void k_edge_alpha(const int* __restrict__ src, const int* __restrict__ dst,
                             const int* __restrict__ et, const float* __restrict__ qn,
                             const float* __restrict__ kn, float* __restrict__ alpha,
                             unsigned* __restrict__ amax, int E, int N) {
    int e = blockIdx.x * blockDim.x + threadIdx.x;
    if (e >= E) return;
    int r = et[e], s = src[e], d = dst[e];
    float a = qn[(size_t)r * N + d] + kn[(size_t)r * N + s];
    a = (a > 0.f) ? a : 0.2f * a;          // leaky_relu(0.2)
    alpha[e] = a;
    atomicMax(&amax[d], fenc(a));
}

__global__ void k_edge_exp(const int* __restrict__ dst, float* __restrict__ alpha,
                           const unsigned* __restrict__ amax, float* __restrict__ denom, int E) {
    int e = blockIdx.x * blockDim.x + threadIdx.x;
    if (e >= E) return;
    int d = dst[e];
    float ex = __expf(alpha[e] - fdec(amax[d]));
    alpha[e] = ex;                          // overwrite alpha with exp
    atomicAdd(&denom[d], ex);
}

// one wave per edge: agg[d] += (ex/denom[d]) * xw[r][s][:]
__global__ void k_edge_agg(const int* __restrict__ src, const int* __restrict__ dst,
                           const int* __restrict__ et, const float* __restrict__ ex,
                           const float* __restrict__ denom, const float* __restrict__ xw,
                           float* __restrict__ agg, int E, int N, int O) {
    int gtid = blockIdx.x * blockDim.x + threadIdx.x;
    int wave = gtid >> 5, lane = gtid & 31;
    if (wave >= E) return;
    int r = et[wave], s = src[wave], d = dst[wave];
    float coef = ex[wave] / (denom[d] + 1e-16f);
    const float* xs = xw + ((size_t)r * N + s) * O;
    float* ad = agg + (size_t)d * O;
    for (int t = lane; t < O; t += 32)
        atomicAdd(&ad[t], coef * xs[t]);
}

__global__ void k_elu_f16(const float* __restrict__ in, float* __restrict__ out32,
                          _Float16* __restrict__ out16, int n) {
    int i = blockIdx.x * blockDim.x + threadIdx.x;
    if (i >= n) return;
    float x = in[i];
    float y = (x > 0.f) ? x : expm1f(x);    // elu(alpha=1)
    out32[i] = y;
    out16[i] = (_Float16)y;
}

// ---------------- launch ----------------
extern "C" void kernel_launch(void* const* d_in, const int* in_sizes, int n_in,
                              void* d_out, int out_size, void* d_ws, size_t ws_size,
                              hipStream_t stream) {
    constexpr int IN = 256, HID = 128, OUT = 64;
    const int N = in_sizes[0] / IN;      // 50000 (multiple of 16)
    const int E = in_sizes[2];           // 800000
    const int R = in_sizes[4] / HID;     // 2
    const int Mt = N / 16;
    const int RG = (Mt + 7) / 8;         // row groups (8 waves per block)

    const float* features = (const float*)d_in[0];
    const int*   src      = (const int*)d_in[1];
    const int*   dst      = src + E;
    const int*   etype    = (const int*)d_in[2];
    const float* W1 = (const float*)d_in[3];
    const float* Q1 = (const float*)d_in[4];
    const float* K1 = (const float*)d_in[5];
    const float* W2 = (const float*)d_in[6];
    const float* Q2 = (const float*)d_in[7];
    const float* K2 = (const float*)d_in[8];
    const float* dW = (const float*)d_in[9];
    const float* db = (const float*)d_in[10];
    float* out = (float*)d_out;          // [h2 (N*OUT) | h3 (N*IN)]

    // workspace layout (bump allocator, 256B aligned, with reuse)
    char* ws = (char*)d_ws;
    size_t off = 0;
    auto alloc = [&](size_t bytes) { size_t o = off; off = (off + bytes + 255) & ~(size_t)255; return o; };

    const int CT1 = HID / 16, KS1 = IN / 32;    // layer1 B tiles
    const int CT2 = OUT / 16, KS2 = HID / 32;   // layer2 B tiles
    const int CTd = IN / 16,  KSd = OUT / 32;   // decoder B tiles
    const size_t f1h = (size_t)CT1 * KS1 * 512; // halves per relation
    const size_t f2h = (size_t)CT2 * KS2 * 512;
    const size_t fdh = (size_t)CTd * KSd * 512;

    size_t o_x16   = alloc((size_t)N * IN * 2);         // xf16; later h1f16 + h2f16
    size_t o_frag1 = alloc((size_t)R * f1h * 2);
    size_t o_frag2 = alloc((size_t)R * f2h * 2);
    size_t o_fragd = alloc(fdh * 2);
    size_t o_xw    = alloc((size_t)R * N * HID * 4);    // xw1; reused for xw2
    size_t o_qn    = alloc((size_t)R * N * 4);
    size_t o_kn    = alloc((size_t)R * N * 4);
    size_t o_amax  = alloc((size_t)N * 4);
    size_t o_denom = alloc((size_t)N * 4);
    size_t o_alpha = alloc((size_t)E * 4);
    size_t o_agg1  = alloc((size_t)N * HID * 4);
    size_t o_agg2  = alloc((size_t)N * OUT * 4);

    _Float16* x16   = (_Float16*)(ws + o_x16);
    _Float16* h1_16 = x16;                                        // reuse after layer1 GEMM
    _Float16* h2_16 = (_Float16*)(ws + o_x16 + (size_t)N * HID * 2);
    _Float16* frag1 = (_Float16*)(ws + o_frag1);
    _Float16* frag2 = (_Float16*)(ws + o_frag2);
    _Float16* fragd = (_Float16*)(ws + o_fragd);
    float*    xw    = (float*)(ws + o_xw);
    float*    qn    = (float*)(ws + o_qn);
    float*    kn    = (float*)(ws + o_kn);
    unsigned* amax  = (unsigned*)(ws + o_amax);
    float*    denom = (float*)(ws + o_denom);
    float*    alpha = (float*)(ws + o_alpha);
    float*    agg1  = (float*)(ws + o_agg1);
    float*    agg2  = (float*)(ws + o_agg2);

    auto nb = [](long long n) { return (int)((n + TPB - 1) / TPB); };

    // --- features -> f16, pack weight fragments
    k_f32_to_f16<<<nb((long long)N * IN), TPB, 0, stream>>>(features, x16, N * IN);
    for (int r = 0; r < R; ++r) {
        k_pack_bfrag<<<nb(CT1 * KS1 * 32), TPB, 0, stream>>>(
            W1 + (size_t)r * IN * HID, frag1 + (size_t)r * f1h, IN, HID);
        k_pack_bfrag<<<nb(CT2 * KS2 * 32), TPB, 0, stream>>>(
            W2 + (size_t)r * HID * OUT, frag2 + (size_t)r * f2h, HID, OUT);
    }
    k_pack_bfrag<<<nb(CTd * KSd * 32), TPB, 0, stream>>>(dW, fragd, OUT, IN);

    // --- layer 1: xw1[r] = X @ W1[r]   (16x64 per wave, B staged in LDS)
    for (int r = 0; r < R; ++r)
        k_wmma_gemm_t<IN, HID / 16, 4><<<RG * (HID / 64), TPB, 0, stream>>>(
            x16, frag1 + (size_t)r * f1h, xw + (size_t)r * N * HID, nullptr, Mt);
    k_node_qk<<<nb((long long)R * N), TPB, 0, stream>>>(xw, Q1, K1, qn, kn, R * N, N, HID);
    hipMemsetAsync(amax, 0, (size_t)N * 4, stream);
    hipMemsetAsync(denom, 0, (size_t)N * 4, stream);
    hipMemsetAsync(agg1, 0, (size_t)N * HID * 4, stream);
    k_edge_alpha<<<nb(E), TPB, 0, stream>>>(src, dst, etype, qn, kn, alpha, amax, E, N);
    k_edge_exp<<<nb(E), TPB, 0, stream>>>(dst, alpha, amax, denom, E);
    k_edge_agg<<<nb((long long)E * 32), TPB, 0, stream>>>(src, dst, etype, alpha, denom,
                                                          xw, agg1, E, N, HID);
    k_elu_f16<<<nb((long long)N * HID), TPB, 0, stream>>>(agg1, agg1, h1_16, N * HID);

    // --- layer 2: xw2[r] = h1 @ W2[r]  (reuses xw buffer)
    for (int r = 0; r < R; ++r)
        k_wmma_gemm_t<HID, OUT / 16, 4><<<RG * (OUT / 64), TPB, 0, stream>>>(
            h1_16, frag2 + (size_t)r * f2h, xw + (size_t)r * N * OUT, nullptr, Mt);
    k_node_qk<<<nb((long long)R * N), TPB, 0, stream>>>(xw, Q2, K2, qn, kn, R * N, N, OUT);
    hipMemsetAsync(amax, 0, (size_t)N * 4, stream);
    hipMemsetAsync(denom, 0, (size_t)N * 4, stream);
    hipMemsetAsync(agg2, 0, (size_t)N * OUT * 4, stream);
    k_edge_alpha<<<nb(E), TPB, 0, stream>>>(src, dst, etype, qn, kn, alpha, amax, E, N);
    k_edge_exp<<<nb(E), TPB, 0, stream>>>(dst, alpha, amax, denom, E);
    k_edge_agg<<<nb((long long)E * 32), TPB, 0, stream>>>(src, dst, etype, alpha, denom,
                                                          xw, agg2, E, N, OUT);
    // h2 = elu(agg2) -> d_out[0:N*OUT], + f16 copy for decoder
    k_elu_f16<<<nb((long long)N * OUT), TPB, 0, stream>>>(agg2, out, h2_16, N * OUT);

    // --- decoder: h3 = h2 @ dec_W + dec_b -> d_out[N*OUT:]
    k_wmma_gemm_t<OUT, IN / 16, 4><<<RG * (IN / 64), TPB, 0, stream>>>(
        h2_16, fragd, out + (size_t)N * OUT, db, Mt);
}
